// Attention_755914244146
// MI455X (gfx1250) — matrix operand order repaired and checked by reading, not compiled, not run on hardware
//
#include <hip/hip_runtime.h>
#include <hip/hip_bf16.h>

typedef __attribute__((ext_vector_type(16))) __bf16 bf16x16;
typedef __attribute__((ext_vector_type(4)))  __bf16 bf16x4;
typedef __attribute__((ext_vector_type(8)))  float  f32x8;
typedef __attribute__((ext_vector_type(4)))  unsigned int u32x4;
typedef __attribute__((ext_vector_type(8)))  int  i32x8;
typedef __attribute__((ext_vector_type(4)))  int  i32x4;

#define DIM    4096
#define NH     32
#define NKV    8
#define HD     128
#define BATCH  2
#define SEQ    2048
#define BS     (BATCH*SEQ)      // 4096 rows
#define KVD    (NKV*HD)         // 1024

#if defined(__has_builtin)
#if __has_builtin(__builtin_amdgcn_tensor_load_to_lds)
#define HAVE_TDM 1
#endif
#endif

// ---------------------------------------------------------------------------
// TDM: async 2D tile load (Global -> LDS) via Tensor Data Mover.
// Builds D# groups per CDNA5 ISA 8.3/8.4 (2D tensor: trailing groups zero).
// 6-arg builtin form (clang-23 / therock headers).
// Issue from ONE wave only (TDM ignores EXEC); caller waits TENSORcnt + barrier.
// ---------------------------------------------------------------------------
#ifdef HAVE_TDM
__device__ __forceinline__ void tdm_load_2d_bf16(const void* gptr, void* lptr,
                                                 int tile_w, int tile_h,
                                                 int stride_elems) {
  unsigned long long ga = (unsigned long long)(uintptr_t)gptr;
  unsigned int lds = (unsigned int)(uintptr_t)lptr;   // flat LDS addr[31:0] = LDS byte offset
  u32x4 g0;
  g0[0] = 1u;                                          // count=1, user mode
  g0[1] = lds;                                         // lds_addr
  g0[2] = (unsigned int)ga;                            // global_addr[31:0]
  g0[3] = (unsigned int)((ga >> 32) & 0x01ffffffu) | (2u << 30); // addr[56:32] | type=2
  i32x8 g1;
  g1[0] = 1 << 16;                                     // data_size=1 (2 bytes)
  g1[1] = (int)(((unsigned)tile_w & 0xffffu) << 16);   // tensor_dim0[15:0] (== tile_w)
  g1[2] = (int)((((unsigned)tile_w >> 16) & 0xffffu) |
                (((unsigned)tile_h & 0xffffu) << 16)); // dim0 hi | tensor_dim1 lo
  g1[3] = (int)((((unsigned)tile_h >> 16) & 0xffffu) |
                (((unsigned)tile_w & 0xffffu) << 16)); // dim1 hi | tile_dim0
  g1[4] = (int)((unsigned)tile_h & 0xffffu);           // tile_dim1 | tile_dim2=0
  g1[5] = stride_elems;                                // tensor_dim0_stride[31:0]
  g1[6] = 0;
  g1[7] = 0;
  i32x4 gz4 = {};
  i32x8 gz8 = {};
  __builtin_amdgcn_tensor_load_to_lds(g0, g1, gz4, gz4, gz8, 0);
}
__device__ __forceinline__ void tdm_wait() {
#if __has_builtin(__builtin_amdgcn_s_wait_tensorcnt)
  __builtin_amdgcn_s_wait_tensorcnt(0);
#else
  asm volatile("s_wait_tensorcnt 0x0" ::: "memory");
#endif
}
#endif

// ---------------------------------------------------------------------------
// f32 -> bf16 elementwise convert
// ---------------------------------------------------------------------------
__global__ void cvt_bf16_kernel(const float* __restrict__ in,
                                __bf16* __restrict__ out, size_t n) {
  size_t i = (size_t)blockIdx.x * blockDim.x + threadIdx.x;
  if (i < n) out[i] = (__bf16)in[i];
}

// ---------------------------------------------------------------------------
// RoPE: read f32 GEMM output, rotate pairs, emit bf16. X: [B*S][nheads*128]
// ---------------------------------------------------------------------------
__global__ void rope_kernel(const float* __restrict__ X,
                            const float* __restrict__ fcos,
                            const float* __restrict__ fsin,
                            __bf16* __restrict__ out, int nheads) {
  int idx = blockIdx.x * blockDim.x + threadIdx.x;
  int total = BATCH * SEQ * nheads * (HD / 2);
  if (idx >= total) return;
  int i = idx & 63;
  int h = (idx >> 6) % nheads;
  int s = (idx / (64 * nheads)) % SEQ;
  int b = idx / (64 * nheads * SEQ);
  size_t base = ((size_t)(b * SEQ + s) * nheads + h) * HD + 2 * i;
  float re = X[base], im = X[base + 1];
  float c = fcos[s * 64 + i], sn = fsin[s * 64 + i];
  out[base]     = (__bf16)(re * c - im * sn);
  out[base + 1] = (__bf16)(re * sn + im * c);
}

// ---------------------------------------------------------------------------
// Tiled WMMA GEMM: C[M,N] (f32) = A[M,K] (bf16) @ B[K,N] (bf16)
// Macro tile 128x128, BK=32, 256 threads = 8 waves as 2(M)x4(N); each wave
// computes 4x2 WMMA tiles (64x32) -> 8 v_wmma per K-step vs 6 fragment loads.
// A tile staged by TDM (async tensor DMA) when available; B staged transposed.
// ---------------------------------------------------------------------------
#define BM 128
#define BN 128
#define BK 32

__global__ __launch_bounds__(256) void gemm_bf16_kernel(
    const __bf16* __restrict__ A, const __bf16* __restrict__ Bm,
    float* __restrict__ C, int M, int N, int K) {
  __shared__ __bf16 sA[BM * BK];      // [128][32] row-major
  __shared__ __bf16 sBT[BN * BK];     // [128][32] transposed (n-major)

  const int tid  = threadIdx.x;
  const int lane = tid & 31;
  const int wid  = tid >> 5;          // 0..7
  const int wm   = wid & 1;           // 2 waves along M (64 rows each)
  const int wn   = wid >> 1;          // 4 waves along N (32 cols each)
  const int m0   = blockIdx.y * BM;
  const int n0   = blockIdx.x * BN;
  const int lr   = lane & 15;
  const int lh   = lane >> 4;

  f32x8 zero = {};
  f32x8 acc[4][2];
  #pragma unroll
  for (int i = 0; i < 4; ++i)
    #pragma unroll
    for (int j = 0; j < 2; ++j) acc[i][j] = zero;

  for (int k0 = 0; k0 < K; k0 += BK) {
#ifdef HAVE_TDM
    if (tid < 32) {                   // wave 0 drives the DMA; TDM ignores EXEC
      tdm_load_2d_bf16(A + (size_t)m0 * K + k0, sA, BK, BM, K);
    }
#else
    { // stage A 128x32 manually
      int row = tid >> 3;             // 0..31
      int col = (tid & 7) * 4;        // 0..28
      #pragma unroll
      for (int it = 0; it < 4; ++it) {
        int r = row + it * 32;
        bf16x4 v = *(const bf16x4*)(A + (size_t)(m0 + r) * K + k0 + col);
        *(bf16x4*)(sA + r * BK + col) = v;
      }
    }
#endif
    { // stage B 32x128, transposed into sBT[n][k]
      int nn = (tid & 31) * 4;        // 0..124
      int kk = tid >> 5;              // 0..7
      #pragma unroll
      for (int it = 0; it < 4; ++it) {
        int k = kk + it * 8;
        bf16x4 v = *(const bf16x4*)(Bm + (size_t)(k0 + k) * N + n0 + nn);
        sBT[(nn + 0) * BK + k] = v[0];
        sBT[(nn + 1) * BK + k] = v[1];
        sBT[(nn + 2) * BK + k] = v[2];
        sBT[(nn + 3) * BK + k] = v[3];
      }
    }
#ifdef HAVE_TDM
    if (tid < 32) tdm_wait();
#endif
    __syncthreads();

    bf16x16 afrag[4], bfrag[2];
    #pragma unroll
    for (int mi = 0; mi < 4; ++mi) {          // A layout: lane r, K = 8h..8h+7, 16+8h..
      const __bf16* ap = sA + (wm * 64 + mi * 16 + lr) * BK;
      #pragma unroll
      for (int i = 0; i < 8; ++i) {
        afrag[mi][i]     = ap[8 * lh + i];
        afrag[mi][8 + i] = ap[16 + 8 * lh + i];
      }
    }
    #pragma unroll
    for (int ni = 0; ni < 2; ++ni) {          // B layout: lane n, K = 8h..8h+7, 16+8h..
      const __bf16* bp = sBT + (wn * 32 + ni * 16 + lr) * BK;
      #pragma unroll
      for (int i = 0; i < 8; ++i) { bfrag[ni][i] = bp[8 * lh + i]; bfrag[ni][8 + i] = bp[16 + 8 * lh + i]; }
    }
    #pragma unroll
    for (int mi = 0; mi < 4; ++mi)
      #pragma unroll
      for (int ni = 0; ni < 2; ++ni)
        acc[mi][ni] = __builtin_amdgcn_wmma_f32_16x16x32_bf16(
            false, afrag[mi], false, bfrag[ni], (short)0, acc[mi][ni], false, false);

    __syncthreads();
  }

  #pragma unroll
  for (int mi = 0; mi < 4; ++mi)
    #pragma unroll
    for (int ni = 0; ni < 2; ++ni)
      #pragma unroll
      for (int j = 0; j < 8; ++j) {           // C layout: row j+8h, col lr
        int row = m0 + wm * 64 + mi * 16 + j + 8 * lh;
        int col = n0 + wn * 32 + ni * 16 + lr;
        C[(size_t)row * N + col] = acc[mi][ni][j];
      }
}

// ---------------------------------------------------------------------------
// Flash attention (causal, GQA). Block = (64 q-rows, head, batch), 4 waves.
// K tile staged via TDM when available; V staged transposed manually.
// ---------------------------------------------------------------------------
__global__ __launch_bounds__(128) void attn_kernel(
    const __bf16* __restrict__ Q,     // [BS][NH*HD]
    const __bf16* __restrict__ Kc,    // [BS][NKV*HD]
    const __bf16* __restrict__ Vc,    // [BS][NKV*HD]
    __bf16* __restrict__ O) {         // [BS][NH*HD]
  __shared__ __bf16 sK[32 * 128];     // K tile, row-major [kpos][d]
  __shared__ __bf16 sVT[128 * 32];    // V tile transposed [d][kpos]
  __shared__ __bf16 sP[4 * 16 * 32];  // per-wave P tiles

  const int tid  = threadIdx.x;
  const int lane = tid & 31;
  const int w    = tid >> 5;          // 0..3
  const int lr   = lane & 15;
  const int lh   = lane >> 4;

  const int qb  = blockIdx.x;         // S/64 = 32
  const int hq  = blockIdx.y;         // 32 heads
  const int bb  = blockIdx.z;         // batch
  const int kvh = hq >> 2;            // GQA: 4 q-heads per kv-head

  const int qrow0 = qb * 64 + w * 16;

  bf16x16 qf[4];
  {
    const __bf16* qp = Q + (size_t)(bb * SEQ + qrow0 + lr) * (NH * HD) + hq * HD;
    #pragma unroll
    for (int s = 0; s < 4; ++s)
      #pragma unroll
      for (int i = 0; i < 8; ++i) {
        qf[s][i]     = qp[s * 32 + 8 * lh + i];
        qf[s][8 + i] = qp[s * 32 + 16 + 8 * lh + i];
      }
  }

  f32x8 zero = {};
  f32x8 acco[8];
  #pragma unroll
  for (int i = 0; i < 8; ++i) acco[i] = zero;
  float mrun[8], lrun[8];
  #pragma unroll
  for (int j = 0; j < 8; ++j) { mrun[j] = -1e30f; lrun[j] = 0.f; }

  const float scale = 0.08838834764831845f;   // 1/sqrt(128)
  const int nkb = qb * 2 + 2;                 // causal: kpos <= qb*64+63

  for (int kb = 0; kb < nkb; ++kb) {
    const int kbase = kb * 32;
#ifdef HAVE_TDM
    if (tid < 32) {
      tdm_load_2d_bf16(Kc + (size_t)(bb * SEQ + kbase) * KVD + kvh * HD,
                       sK, HD, 32, KVD);
    }
#else
    { // stage K 32x128
      int row = tid >> 5;             // 0..3
      int col = (tid & 31) * 4;       // 0..124
      #pragma unroll
      for (int it = 0; it < 8; ++it) {
        int r = row + it * 4;
        bf16x4 v = *(const bf16x4*)(Kc + (size_t)(bb * SEQ + kbase + r) * KVD + kvh * HD + col);
        *(bf16x4*)(sK + r * 128 + col) = v;
      }
    }
#endif
    { // stage V transposed 128x32
      int row = tid >> 5;
      int col = (tid & 31) * 4;
      #pragma unroll
      for (int it = 0; it < 8; ++it) {
        int r = row + it * 4;
        bf16x4 v = *(const bf16x4*)(Vc + (size_t)(bb * SEQ + kbase + r) * KVD + kvh * HD + col);
        sVT[(col + 0) * 32 + r] = v[0];
        sVT[(col + 1) * 32 + r] = v[1];
        sVT[(col + 2) * 32 + r] = v[2];
        sVT[(col + 3) * 32 + r] = v[3];
      }
    }
#ifdef HAVE_TDM
    if (tid < 32) tdm_wait();
#endif
    __syncthreads();

    // scores: 16q x 32k via two 16x16 tiles, contraction over d=128
    f32x8 sc[2];
    #pragma unroll
    for (int ni = 0; ni < 2; ++ni) {
      f32x8 s = zero;
      #pragma unroll
      for (int ks = 0; ks < 4; ++ks) {
        bf16x16 kf;                            // B-frag: col = kpos, K-dim = d
        const __bf16* kp = sK + (ni * 16 + lr) * 128 + ks * 32;
        #pragma unroll
        for (int i = 0; i < 8; ++i) { kf[i] = kp[8 * lh + i]; kf[8 + i] = kp[16 + 8 * lh + i]; }
        s = __builtin_amdgcn_wmma_f32_16x16x32_bf16(
            false, qf[ks], false, kf, (short)0, s, false, false);
      }
      int kpos = kbase + ni * 16 + lr;
      #pragma unroll
      for (int j = 0; j < 8; ++j) {
        int qpos = qrow0 + j + 8 * lh;
        float v = s[j] * scale;
        s[j] = (kpos > qpos) ? -1e30f : v;     // causal mask
      }
      sc[ni] = s;
    }

    // online softmax (row stats live across the 16 lanes of each half)
    float mnew[8], sf[8];
    #pragma unroll
    for (int j = 0; j < 8; ++j) {
      float mt = fmaxf(sc[0][j], sc[1][j]);
      #pragma unroll
      for (int off = 1; off < 16; off <<= 1)
        mt = fmaxf(mt, __shfl_xor(mt, off, 32));
      mnew[j] = fmaxf(mrun[j], mt);
      sf[j]   = __expf(mrun[j] - mnew[j]);
    }
    #pragma unroll
    for (int ni = 0; ni < 2; ++ni)
      #pragma unroll
      for (int j = 0; j < 8; ++j)
        sc[ni][j] = __expf(sc[ni][j] - mnew[j]);
    #pragma unroll
    for (int j = 0; j < 8; ++j) {
      float r = sc[0][j] + sc[1][j];
      #pragma unroll
      for (int off = 1; off < 16; off <<= 1) r += __shfl_xor(r, off, 32);
      lrun[j] = lrun[j] * sf[j] + r;
      mrun[j] = mnew[j];
    }
    #pragma unroll
    for (int dt = 0; dt < 8; ++dt)
      #pragma unroll
      for (int j = 0; j < 8; ++j) acco[dt][j] *= sf[j];

    // C-tile -> A-fragment re-layout of P through per-wave LDS
    __bf16* pw = sP + w * 16 * 32;
    #pragma unroll
    for (int ni = 0; ni < 2; ++ni)
      #pragma unroll
      for (int j = 0; j < 8; ++j)
        pw[(j + 8 * lh) * 32 + ni * 16 + lr] = (__bf16)sc[ni][j];

    bf16x16 pf;
    {
      const __bf16* pp = pw + lr * 32;
      #pragma unroll
      for (int i = 0; i < 8; ++i) {
        pf[i]     = pp[8 * lh + i];
        pf[8 + i] = pp[16 + 8 * lh + i];
      }
    }

    // PV: out 16q x 128d, contraction over kpos=32
    #pragma unroll
    for (int dt = 0; dt < 8; ++dt) {
      bf16x16 vf;
      const __bf16* vp = sVT + (dt * 16 + lr) * 32;
      #pragma unroll
      for (int i = 0; i < 8; ++i) { vf[i] = vp[8 * lh + i]; vf[8 + i] = vp[16 + 8 * lh + i]; }
      acco[dt] = __builtin_amdgcn_wmma_f32_16x16x32_bf16(
          false, pf, false, vf, (short)0, acco[dt], false, false);
    }
    __syncthreads();
  }

  #pragma unroll
  for (int dt = 0; dt < 8; ++dt)
    #pragma unroll
    for (int j = 0; j < 8; ++j) {
      int qpos = qrow0 + j + 8 * lh;
      float o = acco[dt][j] / lrun[j];
      O[(size_t)(bb * SEQ + qpos) * (NH * HD) + hq * HD + dt * 16 + lr] = (__bf16)o;
    }
}

// ---------------------------------------------------------------------------
extern "C" void kernel_launch(void* const* d_in, const int* in_sizes, int n_in,
                              void* d_out, int out_size, void* d_ws, size_t ws_size,
                              hipStream_t stream) {
  (void)in_sizes; (void)n_in; (void)out_size; (void)ws_size;
  const float* x  = (const float*)d_in[0];
  const float* wq = (const float*)d_in[1];
  const float* wk = (const float*)d_in[2];
  const float* wv = (const float*)d_in[3];
  const float* wo = (const float*)d_in[4];
  const float* fc = (const float*)d_in[7];
  const float* fs = (const float*)d_in[8];
  float* out = (float*)d_out;

  char* ws = (char*)d_ws;
  size_t off = 0;
  auto alloc = [&](size_t bytes) {
    void* p = ws + off;
    off += (bytes + 255) & ~(size_t)255;
    return p;
  };
  __bf16* xh  = (__bf16*)alloc((size_t)BS * DIM * 2);
  __bf16* wqh = (__bf16*)alloc((size_t)DIM * DIM * 2);
  __bf16* wkh = (__bf16*)alloc((size_t)DIM * KVD * 2);
  __bf16* wvh = (__bf16*)alloc((size_t)DIM * KVD * 2);
  __bf16* woh = (__bf16*)alloc((size_t)DIM * DIM * 2);
  __bf16* qh  = (__bf16*)alloc((size_t)BS * DIM * 2);
  __bf16* kh  = (__bf16*)alloc((size_t)BS * KVD * 2);
  __bf16* vh  = (__bf16*)alloc((size_t)BS * KVD * 2);
  __bf16* oh  = (__bf16*)alloc((size_t)BS * DIM * 2);
  float*  xqf = (float*)alloc((size_t)BS * DIM * 4);
  float*  xkf = (float*)alloc((size_t)BS * KVD * 4);
  float*  xvf = (float*)alloc((size_t)BS * KVD * 4);

  const int T = 256;
  auto nb = [](size_t n, int t) { return (unsigned)((n + t - 1) / t); };

  // 1) bf16 conversions
  cvt_bf16_kernel<<<nb((size_t)BS * DIM, T), T, 0, stream>>>(x,  xh,  (size_t)BS * DIM);
  cvt_bf16_kernel<<<nb((size_t)DIM * DIM, T), T, 0, stream>>>(wq, wqh, (size_t)DIM * DIM);
  cvt_bf16_kernel<<<nb((size_t)DIM * KVD, T), T, 0, stream>>>(wk, wkh, (size_t)DIM * KVD);
  cvt_bf16_kernel<<<nb((size_t)DIM * KVD, T), T, 0, stream>>>(wv, wvh, (size_t)DIM * KVD);
  cvt_bf16_kernel<<<nb((size_t)DIM * DIM, T), T, 0, stream>>>(wo, woh, (size_t)DIM * DIM);

  // 2) QKV projections (WMMA GEMMs)
  gemm_bf16_kernel<<<dim3(DIM / BN, BS / BM), 256, 0, stream>>>(xh, wqh, xqf, BS, DIM, DIM);
  gemm_bf16_kernel<<<dim3(KVD / BN, BS / BM), 256, 0, stream>>>(xh, wkh, xkf, BS, KVD, DIM);
  gemm_bf16_kernel<<<dim3(KVD / BN, BS / BM), 256, 0, stream>>>(xh, wvh, xvf, BS, KVD, DIM);

  // 3) RoPE (q, k) + V convert -> bf16 (start_pos==0 => cache == fresh K/V)
  rope_kernel<<<nb((size_t)BATCH * SEQ * NH * 64, T), T, 0, stream>>>(xqf, fc, fs, qh, NH);
  rope_kernel<<<nb((size_t)BATCH * SEQ * NKV * 64, T), T, 0, stream>>>(xkf, fc, fs, kh, NKV);
  cvt_bf16_kernel<<<nb((size_t)BS * KVD, T), T, 0, stream>>>(xvf, vh, (size_t)BS * KVD);

  // 4) causal GQA flash attention
  attn_kernel<<<dim3(SEQ / 64, NH, BATCH), 128, 0, stream>>>(qh, kh, vh, oh);

  // 5) output projection -> f32 result
  gemm_bf16_kernel<<<dim3(DIM / BN, BS / BM), 256, 0, stream>>>(oh, woh, out, BS, DIM, DIM);
}
